// SpaceSelfAttentionModality_32169305047478
// MI455X (gfx1250) — matrix operand order repaired and checked by reading, not compiled, not run on hardware
//
#include <hip/hip_runtime.h>

typedef __attribute__((ext_vector_type(16))) __bf16 v16bf;
typedef __attribute__((ext_vector_type(8)))  float  v8f;

#define BT   32
#define SEQ  512
#define DM   768
#define NH   12
#define HD   64
#define NLAT 32
#define MTOT (BT * SEQ)   // 16384
#define LDK32 40          // pitch for K=32 tiles: 80B = 5*16B
#define LDK64 72          // pitch for 64-wide tiles: 144B = 9*16B

// ---------------- fragment helpers (all contiguous 2 x b128) ----------------
// A fragment (16x32 bf16): lane half h holds K = {8h..8h+7, 16+8h..16+8h+7}.
__device__ __forceinline__ v16bf load_afrag(const __bf16* p) {
  union { uint4 u[2]; v16bf v; } c;
  c.u[0] = *(const uint4*)(p);
  c.u[1] = *(const uint4*)(p + 16);
  return c.v;
}
// B fragment (32x16 bf16) with this lane's column K-values contiguous in memory:
// lane half h holds K = {8h..8h+7, 16+8h..16+8h+7}.
__device__ __forceinline__ v16bf load16(const __bf16* p) {
  union { uint4 u[2]; v16bf v; } c;
  c.u[0] = *(const uint4*)(p);
  c.u[1] = *(const uint4*)(p + 16);
  return c.v;
}
__device__ __forceinline__ v8f wmma_bf16(v16bf a, v16bf b, v8f c) {
  return __builtin_amdgcn_wmma_f32_16x16x32_bf16(false, a, false, b, (short)0, c,
                                                 false, false);
}

// ---------------- fp32 -> bf16 convert (vectorized) ----------------
__global__ void cvt4_f32_bf16(const float4* __restrict__ in,
                              __bf16* __restrict__ out, int n4) {
  int i = blockIdx.x * blockDim.x + threadIdx.x;
  int stride = gridDim.x * blockDim.x;
  for (; i < n4; i += stride) {
    float4 f = in[i];
    union { __bf16 h[4]; uint2 u; } c;
    c.h[0] = (__bf16)f.x; c.h[1] = (__bf16)f.y;
    c.h[2] = (__bf16)f.z; c.h[3] = (__bf16)f.w;
    *(uint2*)(out + (size_t)i * 4) = c.u;
  }
}

// ---------------- fp32 -> bf16 transposed convert (DM x DM) ----------------
// out[n*DM + k] = (bf16) in[k*DM + n]
__global__ __launch_bounds__(256) void cvt_transpose(const float* __restrict__ in,
                                                     __bf16* __restrict__ out) {
  __shared__ float tile[32][33];
  int bx = blockIdx.x * 32, by = blockIdx.y * 32;
  int tx = threadIdx.x, ty = threadIdx.y;  // block (32,8)
#pragma unroll
  for (int j = 0; j < 32; j += 8)
    tile[ty + j][tx] = in[(size_t)(by + ty + j) * DM + bx + tx];
  __syncthreads();
#pragma unroll
  for (int j = 0; j < 32; j += 8)
    out[(size_t)(bx + ty + j) * DM + by + tx] = (__bf16)tile[tx][ty + j];
}

// ---------------- QKV projection GEMM (double-buffered) ----------------
// X (MTOT x DM) bf16 * W^T-fed (N x K layout) -> Q/K bf16 (bt,h,s,hd), V bf16
// transposed (bt,h,hd,s). Wave tile 32x64 (8 WMMA / k-step), block tile 128x64.
__global__ __launch_bounds__(128) void qkv_gemm(
    const __bf16* __restrict__ X, const __bf16* __restrict__ WqT,
    const __bf16* __restrict__ WkT, const __bf16* __restrict__ WvT,
    const float* __restrict__ bq, const float* __restrict__ bk,
    const float* __restrict__ bv, __bf16* __restrict__ Qo,
    __bf16* __restrict__ Ko, __bf16* __restrict__ VTo) {
  __shared__ __align__(16) __bf16 sX[2][128 * LDK32];
  __shared__ __align__(16) __bf16 sWT[2][64 * LDK32];

  const int t = threadIdx.x;
  const int wave = t >> 5, lane = t & 31;
  const int half = lane >> 4, col = lane & 15;
  const int mblk = blockIdx.x * 128;
  const int nblk = blockIdx.y * 64;
  const int sel = blockIdx.z;

  const __bf16* WT = (sel == 0) ? WqT : (sel == 1 ? WkT : WvT);
  const float* bias = (sel == 0) ? bq : (sel == 1 ? bk : bv);
  const float scale = (sel == 0) ? 0.125f : 1.0f;

  const int wrow = t >> 1, wcb = (t & 1) * 16;
  const __bf16* Xrow = X + (size_t)(mblk + t) * DM;     // one 32-col slab / thread
  const __bf16* Wrow = WT + (size_t)(nblk + wrow) * DM + wcb;

  uint4 rx[4], rw[2];
  {  // prologue: tile 0 -> regs -> buf0
    const uint4* sx = (const uint4*)(Xrow);
#pragma unroll
    for (int j = 0; j < 4; ++j) rx[j] = sx[j];
    const uint4* sw = (const uint4*)(Wrow);
    rw[0] = sw[0]; rw[1] = sw[1];
    uint4* dx = (uint4*)(&sX[0][t * LDK32]);
#pragma unroll
    for (int j = 0; j < 4; ++j) dx[j] = rx[j];
    uint4* dw = (uint4*)(&sWT[0][wrow * LDK32 + wcb]);
    dw[0] = rw[0]; dw[1] = rw[1];
  }
  __syncthreads();

  v8f acc[2][4] = {};
  const int NIT = DM / 32;
  int cur = 0;
  for (int it = 0; it < NIT; ++it) {
    if (it + 1 < NIT) {  // prefetch next tile into regs (overlaps compute)
      const uint4* sx = (const uint4*)(Xrow + (it + 1) * 32);
#pragma unroll
      for (int j = 0; j < 4; ++j) rx[j] = sx[j];
      const uint4* sw = (const uint4*)(Wrow + (it + 1) * 32);
      rw[0] = sw[0]; rw[1] = sw[1];
    }

    const __bf16* bX = sX[cur];
    const __bf16* bW = sWT[cur];
    v16bf a0 = load_afrag(bX + (wave * 32 + col) * LDK32 + half * 8);
    v16bf a1 = load_afrag(bX + (wave * 32 + 16 + col) * LDK32 + half * 8);
#pragma unroll
    for (int n = 0; n < 4; ++n) {
      v16bf b = load16(bW + (n * 16 + col) * LDK32 + half * 8);
      acc[0][n] = wmma_bf16(a0, b, acc[0][n]);
      acc[1][n] = wmma_bf16(a1, b, acc[1][n]);
    }

    if (it + 1 < NIT) {
      uint4* dx = (uint4*)(&sX[cur ^ 1][t * LDK32]);
#pragma unroll
      for (int j = 0; j < 4; ++j) dx[j] = rx[j];
      uint4* dw = (uint4*)(&sWT[cur ^ 1][wrow * LDK32 + wcb]);
      dw[0] = rw[0]; dw[1] = rw[1];
      __syncthreads();
    }
    cur ^= 1;
  }

#pragma unroll
  for (int ms = 0; ms < 2; ++ms) {
#pragma unroll
    for (int n = 0; n < 4; ++n) {
      int gn = nblk + n * 16 + col;
      int h = gn >> 6, d = gn & 63;
      float bs = bias[gn];
#pragma unroll
      for (int r = 0; r < 8; ++r) {
        int gm = mblk + wave * 32 + ms * 16 + r + 8 * half;
        int bt = gm >> 9, s = gm & 511;
        float v = (acc[ms][n][r] + bs) * scale;
        if (sel == 2)  // V stored transposed per head: (bt,h,hd,s)
          VTo[((size_t)(bt * NH + h) * HD + d) * SEQ + s] = (__bf16)v;
        else if (sel == 1)
          Ko[((size_t)(bt * NH + h) * SEQ + s) * HD + d] = (__bf16)v;
        else
          Qo[((size_t)(bt * NH + h) * SEQ + s) * HD + d] = (__bf16)v;
      }
    }
  }
}

// ---------------- fused masked attention (flash-style, double-buffered) -----
__global__ __launch_bounds__(128) void attn_kernel(
    const __bf16* __restrict__ Q, const __bf16* __restrict__ K,
    const __bf16* __restrict__ VT, const int* __restrict__ modality,
    __bf16* __restrict__ ctx) {
  __shared__ __align__(16) __bf16 sK[2][64 * LDK64];   // keys x hd
  __shared__ __align__(16) __bf16 sVT[2][64 * LDK64];  // hd x keys
  __shared__ __align__(16) __bf16 sP[4 * 16 * LDK64];
  __shared__ int smod[2][64];

  const int t = threadIdx.x;
  const int wave = t >> 5, lane = t & 31;
  const int half = lane >> 4, col = lane & 15;
  const int q0 = blockIdx.x * 64;
  const int h = blockIdx.y;
  const int bt = blockIdx.z;

  const size_t base = (size_t)(bt * NH + h) * SEQ * HD;
  const __bf16* Qb = Q + base;
  const __bf16* Kb = K + base;
  const __bf16* VTb = VT + base;  // (hd, SEQ)

  const int qrow = q0 + wave * 16 + col;
  v16bf qa0 = load_afrag(Qb + (size_t)qrow * HD + half * 8);
  v16bf qa1 = load_afrag(Qb + (size_t)qrow * HD + 32 + half * 8);

  int qmod[8];
  bool qlat[8];
#pragma unroll
  for (int r = 0; r < 8; ++r) {
    int gq = q0 + wave * 16 + r + 8 * half;  // C-slot r of this half
    qlat[r] = (gq < NLAT);
    qmod[r] = modality[gq];
  }

  float m_i[8], l_i[8];
#pragma unroll
  for (int r = 0; r < 8; ++r) { m_i[r] = -1e30f; l_i[r] = 0.0f; }
  v8f acc[4] = {};

  __bf16* sPw = sP + wave * 16 * LDK64;

  const int lrow = t >> 1, lcb = (t & 1) * 32;
  const __bf16* Ksrc = Kb + (size_t)lrow * HD + lcb;
  const __bf16* Vsrc = VTb + (size_t)lrow * SEQ + lcb;

  uint4 rk[4], rv[4];
  int rmod = 0;
  {  // prologue: key-block 0 -> regs -> buf0
    const uint4* pk = (const uint4*)(Ksrc);
    const uint4* pv = (const uint4*)(Vsrc);
#pragma unroll
    for (int j = 0; j < 4; ++j) { rk[j] = pk[j]; rv[j] = pv[j]; }
    if (t < 64) rmod = modality[t];
    uint4* dK = (uint4*)(&sK[0][lrow * LDK64 + lcb]);
    uint4* dV = (uint4*)(&sVT[0][lrow * LDK64 + lcb]);
#pragma unroll
    for (int j = 0; j < 4; ++j) { dK[j] = rk[j]; dV[j] = rv[j]; }
    if (t < 64) smod[0][t] = rmod;
  }
  __syncthreads();

  const int NKB = SEQ / 64;
  int cur = 0;
  for (int it = 0; it < NKB; ++it) {
    if (it + 1 < NKB) {  // prefetch next key-block
      const uint4* pk = (const uint4*)(Ksrc + (size_t)(it + 1) * 64 * HD);
      const uint4* pv = (const uint4*)(Vsrc + (it + 1) * 64);
#pragma unroll
      for (int j = 0; j < 4; ++j) { rk[j] = pk[j]; rv[j] = pv[j]; }
      if (t < 64) rmod = modality[(it + 1) * 64 + t];
    }

    const __bf16* bK = sK[cur];
    const __bf16* bV = sVT[cur];

    // scores S(16x64) = Q * K^T ; B columns are contiguous K rows
    v8f sc[4] = {};
#pragma unroll
    for (int n = 0; n < 4; ++n) {
      v16bf b0 = load16(bK + (n * 16 + col) * LDK64 + half * 8);
      sc[n] = wmma_bf16(qa0, b0, sc[n]);
      v16bf b1 = load16(bK + (n * 16 + col) * LDK64 + 32 + half * 8);
      sc[n] = wmma_bf16(qa1, b1, sc[n]);
    }

    // modality mask
#pragma unroll
    for (int n = 0; n < 4; ++n) {
      int kmod = smod[cur][n * 16 + col];
#pragma unroll
      for (int r = 0; r < 8; ++r) {
        bool ok = qlat[r] || (qmod[r] == kmod);
        sc[n][r] = ok ? sc[n][r] : -1e30f;
      }
    }

    // online softmax; row = 16 lanes of one half x 4 n-subtiles
#pragma unroll
    for (int r = 0; r < 8; ++r) {
      float mx = fmaxf(fmaxf(sc[0][r], sc[1][r]), fmaxf(sc[2][r], sc[3][r]));
#pragma unroll
      for (int off = 1; off < 16; off <<= 1)
        mx = fmaxf(mx, __shfl_xor(mx, off, 32));
      float mn = fmaxf(m_i[r], mx);
      float rescale = __expf(m_i[r] - mn);
      m_i[r] = mn;
      float rowsum = 0.0f;
#pragma unroll
      for (int n = 0; n < 4; ++n) {
        float s = sc[n][r];
        float p = (s > -9.0e29f) ? __expf(s - mn) : 0.0f;
        sc[n][r] = p;
        rowsum += p;
      }
#pragma unroll
      for (int off = 1; off < 16; off <<= 1)
        rowsum += __shfl_xor(rowsum, off, 32);
      l_i[r] = l_i[r] * rescale + rowsum;
#pragma unroll
      for (int n = 0; n < 4; ++n) acc[n][r] *= rescale;
#pragma unroll
      for (int n = 0; n < 4; ++n)
        sPw[(r + 8 * half) * LDK64 + n * 16 + col] = (__bf16)sc[n][r];
    }

    // ctx += P(16x64) * V ; V columns contiguous over keys in sVT
#pragma unroll
    for (int kb2 = 0; kb2 < 2; ++kb2) {
      v16bf pa = load_afrag(sPw + col * LDK64 + kb2 * 32 + half * 8);
#pragma unroll
      for (int n = 0; n < 4; ++n) {
        v16bf vb = load16(bV + (n * 16 + col) * LDK64 + kb2 * 32 + half * 8);
        acc[n] = wmma_bf16(pa, vb, acc[n]);
      }
    }

    if (it + 1 < NKB) {  // stage prefetched key-block into the other buffer
      uint4* dK = (uint4*)(&sK[cur ^ 1][lrow * LDK64 + lcb]);
      uint4* dV = (uint4*)(&sVT[cur ^ 1][lrow * LDK64 + lcb]);
#pragma unroll
      for (int j = 0; j < 4; ++j) { dK[j] = rk[j]; dV[j] = rv[j]; }
      if (t < 64) smod[cur ^ 1][t] = rmod;
      __syncthreads();
    }
    cur ^= 1;
  }

  // normalize + store ctx as (bt*S+q, 768) bf16
#pragma unroll
  for (int n = 0; n < 4; ++n) {
#pragma unroll
    for (int r = 0; r < 8; ++r) {
      int gq = q0 + wave * 16 + r + 8 * half;
      int d = n * 16 + col;
      float v = acc[n][r] / l_i[r];
      ctx[(size_t)(bt * SEQ + gq) * DM + h * HD + d] = (__bf16)v;
    }
  }
}

// ---------------- output projection (W fed transposed, double-buffered) -----
__global__ __launch_bounds__(128) void out_gemm(const __bf16* __restrict__ X,
                                                const __bf16* __restrict__ WT,
                                                const float* __restrict__ bias,
                                                float* __restrict__ Out) {
  __shared__ __align__(16) __bf16 sX[2][128 * LDK32];
  __shared__ __align__(16) __bf16 sWT[2][64 * LDK32];

  const int t = threadIdx.x;
  const int wave = t >> 5, lane = t & 31;
  const int half = lane >> 4, col = lane & 15;
  const int mblk = blockIdx.x * 128;
  const int nblk = blockIdx.y * 64;

  const int wrow = t >> 1, wcb = (t & 1) * 16;
  const __bf16* Xrow = X + (size_t)(mblk + t) * DM;
  const __bf16* Wrow = WT + (size_t)(nblk + wrow) * DM + wcb;

  uint4 rx[4], rw[2];
  {
    const uint4* sx = (const uint4*)(Xrow);
#pragma unroll
    for (int j = 0; j < 4; ++j) rx[j] = sx[j];
    const uint4* sw = (const uint4*)(Wrow);
    rw[0] = sw[0]; rw[1] = sw[1];
    uint4* dx = (uint4*)(&sX[0][t * LDK32]);
#pragma unroll
    for (int j = 0; j < 4; ++j) dx[j] = rx[j];
    uint4* dw = (uint4*)(&sWT[0][wrow * LDK32 + wcb]);
    dw[0] = rw[0]; dw[1] = rw[1];
  }
  __syncthreads();

  v8f acc[2][4] = {};
  const int NIT = DM / 32;
  int cur = 0;
  for (int it = 0; it < NIT; ++it) {
    if (it + 1 < NIT) {
      const uint4* sx = (const uint4*)(Xrow + (it + 1) * 32);
#pragma unroll
      for (int j = 0; j < 4; ++j) rx[j] = sx[j];
      const uint4* sw = (const uint4*)(Wrow + (it + 1) * 32);
      rw[0] = sw[0]; rw[1] = sw[1];
    }

    const __bf16* bX = sX[cur];
    const __bf16* bW = sWT[cur];
    v16bf a0 = load_afrag(bX + (wave * 32 + col) * LDK32 + half * 8);
    v16bf a1 = load_afrag(bX + (wave * 32 + 16 + col) * LDK32 + half * 8);
#pragma unroll
    for (int n = 0; n < 4; ++n) {
      v16bf b = load16(bW + (n * 16 + col) * LDK32 + half * 8);
      acc[0][n] = wmma_bf16(a0, b, acc[0][n]);
      acc[1][n] = wmma_bf16(a1, b, acc[1][n]);
    }

    if (it + 1 < NIT) {
      uint4* dx = (uint4*)(&sX[cur ^ 1][t * LDK32]);
#pragma unroll
      for (int j = 0; j < 4; ++j) dx[j] = rx[j];
      uint4* dw = (uint4*)(&sWT[cur ^ 1][wrow * LDK32 + wcb]);
      dw[0] = rw[0]; dw[1] = rw[1];
      __syncthreads();
    }
    cur ^= 1;
  }

#pragma unroll
  for (int ms = 0; ms < 2; ++ms) {
#pragma unroll
    for (int n = 0; n < 4; ++n) {
      int gn = nblk + n * 16 + col;
      float bs = bias[gn];
#pragma unroll
      for (int r = 0; r < 8; ++r) {
        int gm = mblk + wave * 32 + ms * 16 + r + 8 * half;
        Out[(size_t)gm * DM + gn] = acc[ms][n][r] + bs;
      }
    }
  }
}

// ---------------- launcher ----------------
extern "C" void kernel_launch(void* const* d_in, const int* in_sizes, int n_in,
                              void* d_out, int out_size, void* d_ws, size_t ws_size,
                              hipStream_t stream) {
  const float* x = (const float*)d_in[0];
  const float* Wq = (const float*)d_in[1];
  const float* bq = (const float*)d_in[2];
  const float* Wk = (const float*)d_in[3];
  const float* bk = (const float*)d_in[4];
  const float* Wv = (const float*)d_in[5];
  const float* bv = (const float*)d_in[6];
  const float* Wo = (const float*)d_in[7];
  const float* bo = (const float*)d_in[8];
  const int* modality = (const int*)d_in[9];
  float* out = (float*)d_out;
  (void)in_sizes; (void)n_in; (void)out_size; (void)ws_size;

  char* ws = (char*)d_ws;
  size_t off = 0;
  auto take = [&](size_t bytes) -> char* {
    char* p = ws + off;
    off += (bytes + 255) & ~(size_t)255;
    return p;
  };

  __bf16* xbf = (__bf16*)take((size_t)MTOT * DM * 2);
  __bf16* wqT = (__bf16*)take((size_t)DM * DM * 2);
  __bf16* wkT = (__bf16*)take((size_t)DM * DM * 2);
  __bf16* wvT = (__bf16*)take((size_t)DM * DM * 2);
  __bf16* woT = (__bf16*)take((size_t)DM * DM * 2);
  __bf16* qb = (__bf16*)take((size_t)MTOT * DM * 2);
  __bf16* kb = (__bf16*)take((size_t)MTOT * DM * 2);
  __bf16* vbt = (__bf16*)take((size_t)MTOT * DM * 2);
  __bf16* cb = (__bf16*)take((size_t)MTOT * DM * 2);

  cvt4_f32_bf16<<<2048, 256, 0, stream>>>((const float4*)x, xbf, MTOT * DM / 4);

  dim3 tb(32, 8), tg(DM / 32, DM / 32);
  cvt_transpose<<<tg, tb, 0, stream>>>(Wq, wqT);
  cvt_transpose<<<tg, tb, 0, stream>>>(Wk, wkT);
  cvt_transpose<<<tg, tb, 0, stream>>>(Wv, wvT);
  cvt_transpose<<<tg, tb, 0, stream>>>(Wo, woT);

  dim3 g1(MTOT / 128, DM / 64, 3);
  qkv_gemm<<<g1, 128, 0, stream>>>(xbf, wqT, wkT, wvT, bq, bk, bv, qb, kb, vbt);

  dim3 g2(SEQ / 64, NH, BT);
  attn_kernel<<<g2, 128, 0, stream>>>(qb, kb, vbt, modality, cb);

  dim3 g3(MTOT / 128, DM / 64, 1);
  out_gemm<<<g3, 128, 0, stream>>>(cb, woT, bo, out);
}
